// GeminiMusicModel_38981123179058
// MI455X (gfx1250) — matrix operand order repaired and checked by reading, not compiled, not run on hardware
//
#include <hip/hip_runtime.h>
#include <hip/hip_bf16.h>
#include <math.h>

// ---------------------------------------------------------------------------
// Types
// ---------------------------------------------------------------------------
typedef __bf16 bf16;
typedef __attribute__((ext_vector_type(16))) __bf16 v16bf;
typedef __attribute__((ext_vector_type(8)))  float  v8f;

#define EMB     128
#define NHEAD   8
#define HDIM    16
#define FFN     512
#define NLAYER  6
#define TILE_B  16

// ---------------------------------------------------------------------------
// Parameter bundles (device pointers, passed by value as kernarg)
// Weight matrices are pre-packed in WMMA fragment order:
//   fragment f = kt*(N/16) + nt  holds 32 lanes x 16 contiguous bf16:
//   dst[(f*32 + lane)*16 + i] = W[(kt*32 + (lane>>4)*16 + i)*N + nt*16 + (lane&15)]
// so a B operand is ONE aligned 32-byte load per lane.
// ---------------------------------------------------------------------------
struct LayerP {
  const bf16* qkv_w;  const float* qkv_b;
  const bf16* proj_w; const float* proj_b;
  const bf16* ff_w1;  const float* ff_b1;
  const bf16* ff_w2;  const float* ff_b2;
  const float* ln1_g; const float* ln1_b;
  const float* ln2_g; const float* ln2_b;
};

struct NetP {
  const float* genre_emb; const float* artist_emb;
  const float* np_w1; const float* np_b1; const float* np_ln1g; const float* np_ln1b;
  const bf16*  np_w2; const float* np_b2; const float* np_ln2g; const float* np_ln2b;
  const bf16*  sf_w;  const float* sf_b;  const float* sf_lng;  const float* sf_lnb;
  const bf16*  uf_w;  const float* uf_b;  const float* uf_lng;  const float* uf_lnb;
  LayerP lay[NLAYER];
  const bf16*  sp_w;  const float* sp_b;  const float* sp_lng;  const float* sp_lnb;
  const bf16*  up_w;  const float* up_b;  const float* up_lng;  const float* up_lnb;
  const bf16*  ca_in_w;  const float* ca_in_b;
  const bf16*  ca_out_w; const float* ca_out_b;
  const float* cn_g;  const float* cn_b;
  const bf16*  pr_w1; const float* pr_b1; const float* pr_ln1g; const float* pr_ln1b;
  const bf16*  pr_w2; const float* pr_b2; const float* pr_ln2g; const float* pr_ln2b;
  const bf16*  pr_w3; const float* pr_b3; const float* pr_ln3g; const float* pr_ln3b;
  const float* pr_w4; const float* pr_b4;
};

// ---------------------------------------------------------------------------
// Device helpers
// ---------------------------------------------------------------------------
__device__ __forceinline__ float gelu_f(float x) {
  return 0.5f * x * (1.0f + erff(x * 0.70710678118654752f));
}

// Generic WMMA GEMM: OUT[MT*16 x N] (+)= ACT[MT*16 x K] @ W[K x N] + bias
//   ACT : LDS, f32 (ABF=false) or bf16 (ABF=true), row-major, leading dim ldA
//   Wp  : global bf16 in packed fragment order; ntileFull = (full N of matrix)/16,
//         ntOff/ktOff select a column/row tile window into the full matrix
//   OUT : LDS, f32 (OBF=false) or bf16 (OBF=true), leading dim ldO
//   ACC : accumulate into OUT (f32 only)   DOGELU: apply exact GELU to result
//   MT  : number of 16-row M tiles (1 or 2); B fragment loaded once per (nt,kt)
// K multiple of 32, N multiple of 16. Wave32, EXEC all-1 at call sites.
template <bool ABF, bool OBF, bool ACC, bool DOGELU, int MT>
__device__ void wgemm(const void* actv, int ldA, int K,
                      const bf16* __restrict__ Wp, int ntileFull, int ntOff, int ktOff,
                      const float* bias, int N,
                      void* outv, int ldO, int lane)
{
  const int nlane = lane & 15;           // output column within tile
  const int r0    = (lane >> 4) * 8;     // C/D row base
  const int kbA   = (lane >> 4) * 16;    // A-fragment K base

  for (int nt = 0; nt < (N >> 4); ++nt) {
    v8f c[MT] = {};
    for (int kt = 0; kt < (K >> 5); ++kt) {
      // one 32-byte vector load for the whole B fragment
      const v16bf b = *(const v16bf*)(
          Wp + ((size_t)((ktOff + kt) * ntileFull + ntOff + nt) * 32 + lane) * 16);
#pragma unroll
      for (int m = 0; m < MT; ++m) {
        const int mrow = m * 16 + nlane;
        v16bf a;
        if (ABF) {
          const bf16* pa = (const bf16*)actv + (size_t)mrow * ldA + kt * 32 + kbA;
#pragma unroll
          for (int i = 0; i < 8; ++i) { a[i] = pa[i]; a[8 + i] = pa[8 + i]; }
        } else {
          const float* pa = (const float*)actv + (size_t)mrow * ldA + kt * 32 + kbA;
#pragma unroll
          for (int i = 0; i < 8; ++i) { a[i] = (bf16)pa[i]; a[8 + i] = (bf16)pa[8 + i]; }
        }
        c[m] = __builtin_amdgcn_wmma_f32_16x16x32_bf16(false, a, false, b,
                                                       (short)0, c[m], false, false);
      }
    }
    const float bb = bias ? bias[nt * 16 + nlane] : 0.0f;
#pragma unroll
    for (int m = 0; m < MT; ++m) {
#pragma unroll
      for (int r = 0; r < 8; ++r) {
        float v = c[m][r] + bb;
        if (DOGELU) v = gelu_f(v);
        const int orow = m * 16 + r0 + r;
        if (OBF) {
          ((bf16*)outv)[(size_t)orow * ldO + nt * 16 + nlane] = (bf16)v;
        } else {
          float* o = (float*)outv + (size_t)orow * ldO + nt * 16 + nlane;
          if (ACC) *o += v; else *o = v;
        }
      }
    }
  }
}

// Row-wise LayerNorm over LDS buffer [rows x W] (leading dim ld)
__device__ void ln_rows(float* buf, int ld, int rows, int W,
                        const float* g, const float* b, int lane)
{
  for (int r = lane; r < rows; r += 32) {
    float* row = buf + (size_t)r * ld;
    float m = 0.f;
    for (int i = 0; i < W; ++i) m += row[i];
    m /= (float)W;
    float v = 0.f;
    for (int i = 0; i < W; ++i) { float d = row[i] - m; v += d * d; }
    v /= (float)W;
    const float inv = rsqrtf(v + 1e-5f);
    for (int i = 0; i < W; ++i) row[i] = (row[i] - m) * inv * g[i] + b[i];
  }
}

__device__ __forceinline__ void gelu_buf(float* buf, int n, int lane) {
  for (int i = lane; i < n; i += 32) buf[i] = gelu_f(buf[i]);
}

// ---------------------------------------------------------------------------
// Weight repack kernel: f32 row-major [K][N] -> bf16 WMMA fragment order
// ---------------------------------------------------------------------------
__global__ __launch_bounds__(256) void repack_w_k(const float* __restrict__ src,
                                                  bf16* __restrict__ dst,
                                                  int K, int N)
{
  const int i = blockIdx.x * 256 + threadIdx.x;
  if (i >= K * N) return;
  const int e    = i & 15;
  const int lane = (i >> 4) & 31;
  const int f    = i >> 9;
  const int ntile = N >> 4;
  const int kt = f / ntile, nt = f - kt * ntile;
  const int krow = kt * 32 + (lane >> 4) * 16 + e;
  const int col  = nt * 16 + (lane & 15);
  dst[i] = (bf16)src[(size_t)krow * N + col];
}

// ---------------------------------------------------------------------------
// Fused forward kernel: 1 wave per block, 16 batch rows per block.
// LDS: XF (f32 32x128), SC (f32 32x128), HB (bf16 32x384) = 56 KB.
// Sequence row layout: row = s*16 + b  (s=0 song token, s=1 user token).
// ---------------------------------------------------------------------------
__global__ __launch_bounds__(32) void music_forward(
    NetP p,
    const int* __restrict__ sgenre, const int* __restrict__ sartist,
    const int* __restrict__ ugenre, const int* __restrict__ uartist,
    const float* __restrict__ wpm, const float* __restrict__ dur,
    const float* __restrict__ tok, float* __restrict__ out)
{
  __shared__ float XF[32 * 128];   // residual stream / f32 staging
  __shared__ float SC[32 * 128];   // f32 scratch
  __shared__ bf16  HB[32 * 384];   // bf16 staging (qkv / ff halves / concat)

  const int lane = threadIdx.x;
  const int b0   = blockIdx.x * TILE_B;

  // ---- numeric path: [16,3] @ np_w1 -> [16,64] ----
  for (int idx = lane; idx < 16 * 64; idx += 32) {
    const int r = idx >> 6, cc = idx & 63;
    const int gb = b0 + r;
    SC[r * 64 + cc] = wpm[gb] * p.np_w1[0 * 64 + cc]
                    + dur[gb] * p.np_w1[1 * 64 + cc]
                    + tok[gb] * p.np_w1[2 * 64 + cc] + p.np_b1[cc];
  }
  __syncthreads();
  ln_rows(SC, 64, 16, 64, p.np_ln1g, p.np_ln1b, lane);
  gelu_buf(SC, 16 * 64, lane);
  __syncthreads();
  // num_vec = h @ np_w2 + b2 -> XF rows 0-15 as [16][128]
  wgemm<false, false, false, false, 1>(SC, 64, 64, p.np_w2, 8, 0, 0,
                                       p.np_b2, 128, XF, 128, lane);
  __syncthreads();
  ln_rows(XF, 128, 16, 128, p.np_ln2g, p.np_ln2b, lane);
  gelu_buf(XF, 16 * 128, lane);
  __syncthreads();

  // ---- song fusion: concat(sg, sa, num_vec) [16,384] @ sf_w ----
  for (int idx = lane; idx < 16 * 384; idx += 32) {
    const int r = idx / 384, c = idx % 384;
    float v;
    if (c < 128)       v = p.genre_emb[(size_t)sgenre[b0 + r] * 128 + c];
    else if (c < 256)  v = p.artist_emb[(size_t)sartist[b0 + r] * 128 + (c - 128)];
    else               v = XF[r * 128 + (c - 256)];
    HB[idx] = (bf16)v;
  }
  __syncthreads();
  wgemm<true, false, false, false, 1>(HB, 384, 384, p.sf_w, 8, 0, 0,
                                      p.sf_b, 128, SC, 128, lane);
  __syncthreads();
  ln_rows(SC, 128, 16, 128, p.sf_lng, p.sf_lnb, lane);
  gelu_buf(SC, 16 * 128, lane);
  __syncthreads();

  // ---- user fusion: concat(ug, ua) [16,256] @ uf_w ----
  for (int idx = lane; idx < 16 * 256; idx += 32) {
    const int r = idx >> 8, c = idx & 255;
    float v;
    if (c < 128) v = p.genre_emb[(size_t)ugenre[b0 + r] * 128 + c];
    else         v = p.artist_emb[(size_t)uartist[b0 + r] * 128 + (c - 128)];
    HB[idx] = (bf16)v;
  }
  __syncthreads();
  wgemm<true, false, false, false, 1>(HB, 256, 256, p.uf_w, 8, 0, 0,
                                      p.uf_b, 128, SC + 16 * 128, 128, lane);
  __syncthreads();
  ln_rows(SC + 16 * 128, 128, 16, 128, p.uf_lng, p.uf_lnb, lane);
  gelu_buf(SC + 16 * 128, 16 * 128, lane);
  __syncthreads();

  // residual stream XF[32][128] = stacked sequence
  for (int idx = lane; idx < 32 * 128; idx += 32) XF[idx] = SC[idx];
  __syncthreads();

  // ---- 6 transformer blocks ----
  for (int li = 0; li < NLAYER; ++li) {
    const LayerP& L = p.lay[li];

    // qkv: [32,128] @ [128,384] -> HB bf16 [32][384]
    wgemm<false, true, false, false, 2>(XF, 128, 128, L.qkv_w, 24, 0, 0,
                                        L.qkv_b, 384, HB, 384, lane);
    __syncthreads();

    // 2x2 attention per (batch, head) -> SC f32 [32][128]
    for (int t = lane; t < 16 * NHEAD; t += 32) {
      const int b = t & 15, h = t >> 4;
      const int cq = h * HDIM, ck = 128 + h * HDIM, cv = 256 + h * HDIM;
      const bf16* rp0 = HB + (size_t)(b) * 384;        // s=0 row
      const bf16* rp1 = HB + (size_t)(16 + b) * 384;   // s=1 row
      float s00 = 0.f, s01 = 0.f, s10 = 0.f, s11 = 0.f;
      for (int d = 0; d < HDIM; ++d) {
        const float q0 = (float)rp0[cq + d], q1 = (float)rp1[cq + d];
        const float k0 = (float)rp0[ck + d], k1 = (float)rp1[ck + d];
        s00 += q0 * k0; s01 += q0 * k1; s10 += q1 * k0; s11 += q1 * k1;
      }
      s00 *= 0.25f; s01 *= 0.25f; s10 *= 0.25f; s11 *= 0.25f;   // HD^-0.5
      const float m0 = fmaxf(s00, s01), m1 = fmaxf(s10, s11);
      const float e00 = expf(s00 - m0), e01 = expf(s01 - m0);
      const float e10 = expf(s10 - m1), e11 = expf(s11 - m1);
      const float i0 = 1.0f / (e00 + e01), i1 = 1.0f / (e10 + e11);
      const float p00 = e00 * i0, p01 = e01 * i0;
      const float p10 = e10 * i1, p11 = e11 * i1;
      for (int d = 0; d < HDIM; ++d) {
        const float v0 = (float)rp0[cv + d], v1 = (float)rp1[cv + d];
        SC[(size_t)(b) * 128 + h * HDIM + d]      = p00 * v0 + p01 * v1;
        SC[(size_t)(16 + b) * 128 + h * HDIM + d] = p10 * v0 + p11 * v1;
      }
    }
    __syncthreads();

    // proj with residual: XF += SC @ proj_w + proj_b ; then LN1
    wgemm<false, false, true, false, 2>(SC, 128, 128, L.proj_w, 8, 0, 0,
                                        L.proj_b, 128, XF, 128, lane);
    __syncthreads();
    ln_rows(XF, 128, 32, 128, L.ln1_g, L.ln1_b, lane);
    __syncthreads();

    // FF: SC = ff_b2 ; two 256-col halves streamed through HB
    for (int idx = lane; idx < 32 * 128; idx += 32) SC[idx] = L.ff_b2[idx & 127];
    __syncthreads();
    for (int half = 0; half < 2; ++half) {
      // h1 half = gelu(XF @ ff_w1[:, half*256 +: 256] + b1) -> HB bf16 [32][256]
      wgemm<false, true, false, true, 2>(XF, 128, 128, L.ff_w1, 32, half * 16, 0,
                                         L.ff_b1 + half * 256, 256, HB, 256, lane);
      __syncthreads();
      // SC += h1half @ ff_w2[half*256 +: 256, :]
      wgemm<true, false, true, false, 2>(HB, 256, 256, L.ff_w2, 8, 0, half * 8,
                                         nullptr, 128, SC, 128, lane);
      __syncthreads();
    }
    // x = LN2(x + ff)
    for (int idx = lane; idx < 32 * 128; idx += 32) XF[idx] += SC[idx];
    __syncthreads();
    ln_rows(XF, 128, 32, 128, L.ln2_g, L.ln2_b, lane);
    __syncthreads();
  }

  // ---- heads ----
  wgemm<false, false, false, false, 1>(XF, 128, 128, p.sp_w, 8, 0, 0,
                                       p.sp_b, 128, SC, 128, lane);
  wgemm<false, false, false, false, 1>(XF + 16 * 128, 128, 128, p.up_w, 8, 0, 0,
                                       p.up_b, 128, SC + 16 * 128, 128, lane);
  __syncthreads();
  ln_rows(SC, 128, 16, 128, p.sp_lng, p.sp_lnb, lane);
  ln_rows(SC + 16 * 128, 128, 16, 128, p.up_lng, p.up_lnb, lane);
  gelu_buf(SC, 32 * 128, lane);
  __syncthreads();

  // vproj = sp @ ca_in_w[:, 256:384] + ca_in_b[256:384] -> XF rows 0-15
  wgemm<false, false, false, false, 1>(SC, 128, 128, p.ca_in_w, 24, 16, 0,
                                       p.ca_in_b + 256, 128, XF, 128, lane);
  __syncthreads();
  // co = vproj @ ca_out_w + b -> XF rows 16-31
  wgemm<false, false, false, false, 1>(XF, 128, 128, p.ca_out_w, 8, 0, 0,
                                       p.ca_out_b, 128, XF + 16 * 128, 128, lane);
  __syncthreads();
  // co = ln(co + up, cn)
  for (int idx = lane; idx < 16 * 128; idx += 32)
    XF[16 * 128 + idx] += SC[16 * 128 + idx];
  __syncthreads();
  ln_rows(XF + 16 * 128, 128, 16, 128, p.cn_g, p.cn_b, lane);
  __syncthreads();

  // c = concat(sp, co) -> HB bf16 [16][256]
  for (int idx = lane; idx < 16 * 256; idx += 32) {
    const int r = idx >> 8, c = idx & 255;
    const float v = (c < 128) ? SC[r * 128 + c]
                              : XF[16 * 128 + r * 128 + (c - 128)];
    HB[idx] = (bf16)v;
  }
  __syncthreads();

  // pr1: [16,256]@[256,256] -> XF viewed as [16][256]
  wgemm<true, false, false, false, 1>(HB, 256, 256, p.pr_w1, 16, 0, 0,
                                      p.pr_b1, 256, XF, 256, lane);
  __syncthreads();
  ln_rows(XF, 256, 16, 256, p.pr_ln1g, p.pr_ln1b, lane);
  gelu_buf(XF, 16 * 256, lane);
  __syncthreads();

  // pr2: [16,256]@[256,128] -> SC [16][128]
  wgemm<false, false, false, false, 1>(XF, 256, 256, p.pr_w2, 8, 0, 0,
                                       p.pr_b2, 128, SC, 128, lane);
  __syncthreads();
  ln_rows(SC, 128, 16, 128, p.pr_ln2g, p.pr_ln2b, lane);
  gelu_buf(SC, 16 * 128, lane);
  __syncthreads();

  // pr3: [16,128]@[128,64] -> XF [16][64]
  wgemm<false, false, false, false, 1>(SC, 128, 128, p.pr_w3, 4, 0, 0,
                                       p.pr_b3, 64, XF, 64, lane);
  __syncthreads();
  ln_rows(XF, 64, 16, 64, p.pr_ln3g, p.pr_ln3b, lane);
  gelu_buf(XF, 16 * 64, lane);
  __syncthreads();

  // pr4 + sigmoid
  if (lane < 16) {
    float acc = p.pr_b4[0];
    for (int i = 0; i < 64; ++i) acc += XF[lane * 64 + i] * p.pr_w4[i];
    out[b0 + lane] = 1.0f / (1.0f + expf(-acc));
  }
}

// ---------------------------------------------------------------------------
// Host launcher
// ---------------------------------------------------------------------------
extern "C" void kernel_launch(void* const* d_in, const int* in_sizes, int n_in,
                              void* d_out, int out_size, void* d_ws, size_t ws_size,
                              hipStream_t stream)
{
  (void)n_in; (void)ws_size; (void)out_size;
  const int B = in_sizes[0];

  // d_in layout: 7 input arrays, then params leaves in dict insertion order.
  const int* sgenre  = (const int*)d_in[0];
  const int* sartist = (const int*)d_in[1];
  const int* ugenre  = (const int*)d_in[2];
  const int* uartist = (const int*)d_in[3];
  const float* wpm = (const float*)d_in[4];
  const float* dur = (const float*)d_in[5];
  const float* tok = (const float*)d_in[6];

  auto F = [&](int i) { return (const float*)d_in[i]; };

  // bump allocator over workspace; repack f32 [K][N] -> bf16 fragment order
  size_t off = 0;
  auto rpk = [&](int idx, int K, int N) -> const bf16* {
    const int n = K * N;
    bf16* dst = (bf16*)((char*)d_ws + off);
    off += (((size_t)n * sizeof(bf16)) + 255) & ~(size_t)255;
    repack_w_k<<<dim3((n + 255) / 256), dim3(256), 0, stream>>>(F(idx), dst, K, N);
    return dst;
  };

  NetP p;
  p.genre_emb = F(7);   p.artist_emb = F(8);
  p.np_w1 = F(9);  p.np_b1 = F(10); p.np_ln1g = F(11); p.np_ln1b = F(12);
  p.np_w2 = rpk(13, 64, 128);
  p.np_b2 = F(14); p.np_ln2g = F(15); p.np_ln2b = F(16);
  p.sf_w  = rpk(17, 384, 128);
  p.sf_b  = F(18); p.sf_lng = F(19); p.sf_lnb = F(20);
  p.uf_w  = rpk(21, 256, 128);
  p.uf_b  = F(22); p.uf_lng = F(23); p.uf_lnb = F(24);

  for (int li = 0; li < NLAYER; ++li) {
    const int base = 25 + li * 12;
    LayerP& L = p.lay[li];
    L.qkv_w  = rpk(base + 0, 128, 384);  L.qkv_b  = F(base + 1);
    L.proj_w = rpk(base + 2, 128, 128);  L.proj_b = F(base + 3);
    L.ff_w1  = rpk(base + 4, 128, 512);  L.ff_b1  = F(base + 5);
    L.ff_w2  = rpk(base + 6, 512, 128);  L.ff_b2  = F(base + 7);
    L.ln1_g  = F(base + 8);  L.ln1_b = F(base + 9);
    L.ln2_g  = F(base + 10); L.ln2_b = F(base + 11);
  }

  p.sp_w = rpk(97, 128, 128);  p.sp_b = F(98);  p.sp_lng = F(99);  p.sp_lnb = F(100);
  p.up_w = rpk(101, 128, 128); p.up_b = F(102); p.up_lng = F(103); p.up_lnb = F(104);
  p.ca_in_w  = rpk(105, 128, 384); p.ca_in_b  = F(106);
  p.ca_out_w = rpk(107, 128, 128); p.ca_out_b = F(108);
  p.cn_g = F(109); p.cn_b = F(110);
  p.pr_w1 = rpk(111, 256, 256); p.pr_b1 = F(112); p.pr_ln1g = F(113); p.pr_ln1b = F(114);
  p.pr_w2 = rpk(115, 256, 128); p.pr_b2 = F(116); p.pr_ln2g = F(117); p.pr_ln2b = F(118);
  p.pr_w3 = rpk(119, 128, 64);  p.pr_b3 = F(120); p.pr_ln3g = F(121); p.pr_ln3b = F(122);
  p.pr_w4 = F(123); p.pr_b4 = F(124);

  const int nblk = B / TILE_B;
  music_forward<<<dim3(nblk), dim3(32), 0, stream>>>(
      p, sgenre, sartist, ugenre, uartist, wpm, dur, tok, (float*)d_out);
}
